// TernaryMLP_31688268710666
// MI455X (gfx1250) — matrix operand order repaired and checked by reading, not compiled, not run on hardware
//
#include <hip/hip_runtime.h>
#include <hip/hip_bf16.h>
#include <stdint.h>

typedef __attribute__((ext_vector_type(16))) _Float16 v16h;
typedef __attribute__((ext_vector_type(8)))  _Float16 v8h;
typedef __attribute__((ext_vector_type(4)))  _Float16 v4h;
typedef __attribute__((ext_vector_type(8)))  float    v8f;
typedef __attribute__((ext_vector_type(4)))  uint32_t u32x4;
typedef __attribute__((ext_vector_type(8)))  uint32_t u32x8;

#define D 4096
#define NTOK 8192
#define WELEM (4096UL * 4096UL)
#define RED_BLOCKS 512

// ---------------------------------------------------------------------------
// 1) per-block partial sums of |w|  (deterministic: fixed partial slots)
// ---------------------------------------------------------------------------
__global__ __launch_bounds__(256) void absmean_kernel(const float* __restrict__ w,
                                                      float* __restrict__ partials,
                                                      int n) {
  __shared__ float red[256];
  int t = threadIdx.x;
  float s = 0.0f;
  for (int i = blockIdx.x * 256 + t; i < n; i += gridDim.x * 256)
    s += fabsf(w[i]);
  red[t] = s;
  __syncthreads();
  for (int k = 128; k > 0; k >>= 1) {
    if (t < k) red[t] += red[t + k];
    __syncthreads();
  }
  if (t == 0) partials[blockIdx.x] = red[0];
}

// ---------------------------------------------------------------------------
// 2) deterministic final reduction of 512 partials -> abs-sum
// ---------------------------------------------------------------------------
__global__ __launch_bounds__(256) void finalize_scale_kernel(const float* __restrict__ partials,
                                                             float* __restrict__ scale_out) {
  __shared__ float red[256];
  int t = threadIdx.x;
  red[t] = partials[t] + partials[t + 256];
  __syncthreads();
  for (int k = 128; k > 0; k >>= 1) {
    if (t < k) red[t] += red[t + k];
    __syncthreads();
  }
  if (t == 0) scale_out[0] = red[0];
}

// ---------------------------------------------------------------------------
// 3) ternarize: clip(round(w/scale), -1, 1) -> f16 {-1,0,1}
//    rintf == round-half-to-even, matching jnp.round
// ---------------------------------------------------------------------------
__global__ __launch_bounds__(256) void ternarize_kernel(const float* __restrict__ w,
                                                        const float* __restrict__ abs_sum,
                                                        _Float16* __restrict__ out,
                                                        int n4) {
  int i = blockIdx.x * 256 + threadIdx.x;
  if (i >= n4) return;
  float scale = abs_sum[0] * (1.0f / 16777216.0f) + 1e-8f;
  float inv = 1.0f / scale;
  float4 v = ((const float4*)w)[i];
  v4h q;
  q[0] = (_Float16)fminf(1.0f, fmaxf(-1.0f, rintf(v.x * inv)));
  q[1] = (_Float16)fminf(1.0f, fmaxf(-1.0f, rintf(v.y * inv)));
  q[2] = (_Float16)fminf(1.0f, fmaxf(-1.0f, rintf(v.z * inv)));
  q[3] = (_Float16)fminf(1.0f, fmaxf(-1.0f, rintf(v.w * inv)));
  *(v4h*)(out + (size_t)i * 4) = q;
}

// ---------------------------------------------------------------------------
// 4) RMSNorm (eps = FLT_EPSILON) -> SiLU -> f16, one 256-thread block per row
// ---------------------------------------------------------------------------
__global__ __launch_bounds__(256) void norm_silu_kernel(const float* __restrict__ in,
                                                        _Float16* __restrict__ out) {
  __shared__ float red[256];
  const int row = blockIdx.x;
  const int t = threadIdx.x;
  const float* xr = in + (size_t)row * D;
  _Float16* orow = out + (size_t)row * D;

  float v[16];
  float ss = 0.0f;
#pragma unroll
  for (int j = 0; j < 4; ++j) {
    float4 f = *(const float4*)(xr + t * 16 + j * 4);
    v[j * 4 + 0] = f.x; v[j * 4 + 1] = f.y; v[j * 4 + 2] = f.z; v[j * 4 + 3] = f.w;
    ss += f.x * f.x + f.y * f.y + f.z * f.z + f.w * f.w;
  }
  red[t] = ss;
  __syncthreads();
  for (int k = 128; k > 0; k >>= 1) {
    if (t < k) red[t] += red[t + k];
    __syncthreads();
  }
  float ms = red[0] * (1.0f / (float)D);
  float rs = rsqrtf(ms + 1.1920929e-7f);

#pragma unroll
  for (int j = 0; j < 2; ++j) {
    v8h h;
#pragma unroll
    for (int e = 0; e < 8; ++e) {
      float hv = v[j * 8 + e] * rs;
      float s  = hv / (1.0f + __expf(-hv));   // SiLU
      h[e] = (_Float16)s;
    }
    *(v8h*)(orow + t * 16 + j * 8) = h;
  }
}

// ---------------------------------------------------------------------------
// 5) GEMM: C[M,N] = A[M,K] * W[N,K]^T  (A,W f16, C f32)
//    128x128 tile / block (256 threads = 8 wave32, each wave 32x64).
//    K staged in 64-wide tiles, double-buffered in LDS, filled by the
//    Tensor Data Mover (tensor_load_to_lds), completion via TENSORcnt.
// ---------------------------------------------------------------------------
#define TM 128
#define TN 128
#define TK 64
#define LDS_PITCH 72   // 64 + 8 halves pad, produced by TDM pad_interval/amount

// Issue one TDM 2-D tile load: 128 rows x 64 f16, row stride 4096 elements,
// LDS destination padded 4 DWORDs after every 32 DWORDs (row pitch 144 B).
// D# group0: count=1, lds_addr, global_addr, type=2. group1: data_size=2B,
// pad_enable, pad_interval=32dw, pad_amount=4dw, dims/strides/tile dims.
__device__ __forceinline__ void tdm_load_tile(uint32_t lds_off,
                                              const _Float16* gptr,
                                              uint32_t rows) {
  uint64_t ga = (uint64_t)(uintptr_t)gptr;
  u32x4 g0 = { 1u,                                   // count=1 (valid D#)
               lds_off,                              // lds_addr[31:0]
               (uint32_t)ga,                         // global_addr[31:0]
               ((uint32_t)(ga >> 32) & 0x01FFFFFFu)  // global_addr[56:32]
                   | 0x80000000u };                  // type=2 ("image")
  u32x8 g1 = { 0x07110000u,      // data_size=1(2B) pad_en pad_int=32dw pad_amt=4dw
               0x10000000u,      // tensor_dim0 = 4096  (bits[79:48], low part)
               rows << 16,       // tensor_dim0 hi=0 | tensor_dim1 = rows
               64u << 16,        // tensor_dim1 hi=0 | tile_dim0 = 64
               128u,             // tile_dim1 = 128, tile_dim2 = 0
               4096u,            // tensor_dim0_stride[31:0] = 4096 elements
               0u,               // stride hi = 0 | tensor_dim1_stride lo = 0
               0u };             // tensor_dim1_stride hi = 0
  asm volatile("tensor_load_to_lds %0, %1" :: "s"(g0), "s"(g1) : "memory");
}

__global__ __launch_bounds__(256) void gemm_tern_kernel(const _Float16* __restrict__ A,
                                                        const _Float16* __restrict__ W,
                                                        float* __restrict__ C) {
  __shared__ _Float16 As[2][TM][LDS_PITCH];
  __shared__ _Float16 Bs[2][TN][LDS_PITCH];

  const int t = threadIdx.x;
  const int lane = t & 31;
  const int wv = t >> 5;
  const int waveM = wv & 3;        // 4 waves over M (32 rows each)
  const int waveN = wv >> 2;       // 2 waves over N (64 cols each)
  const int lm = lane & 15;
  const int hi = lane >> 4;        // lane group selects K-half per ISA layout

  const int mBase = blockIdx.y * TM;
  const int nBase = blockIdx.x * TN;

  const uint32_t ldsA[2] = { (uint32_t)(uintptr_t)&As[0][0][0],
                             (uint32_t)(uintptr_t)&As[1][0][0] };
  const uint32_t ldsB[2] = { (uint32_t)(uintptr_t)&Bs[0][0][0],
                             (uint32_t)(uintptr_t)&Bs[1][0][0] };
  const _Float16* aTile = A + (size_t)mBase * D;   // uniform per workgroup
  const _Float16* wTile = W + (size_t)nBase * D;

  v8f acc[2][4];
#pragma unroll
  for (int mi = 0; mi < 2; ++mi)
#pragma unroll
    for (int ni = 0; ni < 4; ++ni)
      acc[mi][ni] = (v8f){0.f, 0.f, 0.f, 0.f, 0.f, 0.f, 0.f, 0.f};

  const int KT = D / TK;   // 64 k-tiles

  // prologue: DMA tile 0 into buffer 0
  if (wv == 0) {
    tdm_load_tile(ldsA[0], aTile, (uint32_t)NTOK);
    tdm_load_tile(ldsB[0], wTile, (uint32_t)D);
    __builtin_amdgcn_s_wait_tensorcnt(0);
  }
  __syncthreads();

  for (int kt = 0; kt < KT; ++kt) {
    const int cur = kt & 1;

    // DMA next k-tile into the other buffer while we compute this one
    if (wv == 0 && kt + 1 < KT) {
      tdm_load_tile(ldsA[cur ^ 1], aTile + (size_t)(kt + 1) * TK, (uint32_t)NTOK);
      tdm_load_tile(ldsB[cur ^ 1], wTile + (size_t)(kt + 1) * TK, (uint32_t)D);
    }

#pragma unroll
    for (int kh = 0; kh < 2; ++kh) {
      // A fragments: lane(lm)=M row; VGPR0-3 K=hi*8.., VGPR4-7 K=16+hi*8..
      v16h af[2];
#pragma unroll
      for (int mi = 0; mi < 2; ++mi) {
        const _Float16* ap = &As[cur][waveM * 32 + mi * 16 + lm][kh * 32];
        v8h lo = *(const v8h*)(ap + hi * 8);
        v8h hc = *(const v8h*)(ap + 16 + hi * 8);
        af[mi] = __builtin_shufflevector(lo, hc, 0, 1, 2, 3, 4, 5, 6, 7,
                                                 8, 9, 10, 11, 12, 13, 14, 15);
      }
      // B fragments: lane(lm)=N col=W row; VGPR0-3 K=hi*8.., VGPR4-7 K=16+hi*8..
      v16h bf[4];
#pragma unroll
      for (int ni = 0; ni < 4; ++ni) {
        const _Float16* bp = &Bs[cur][waveN * 64 + ni * 16 + lm][kh * 32];
        v8h blo = *(const v8h*)(bp + hi * 8);
        v8h bhi = *(const v8h*)(bp + 16 + hi * 8);
        bf[ni] = __builtin_shufflevector(blo, bhi, 0, 1, 2, 3, 4, 5, 6, 7,
                                                   8, 9, 10, 11, 12, 13, 14, 15);
      }
#pragma unroll
      for (int mi = 0; mi < 2; ++mi)
#pragma unroll
        for (int ni = 0; ni < 4; ++ni)
          acc[mi][ni] = __builtin_amdgcn_wmma_f32_16x16x32_f16(
              false, af[mi], false, bf[ni], (short)0, acc[mi][ni], false, false);
    }

    if (wv == 0) __builtin_amdgcn_s_wait_tensorcnt(0);
    __syncthreads();
  }

  // epilogue: C/D layout — VGPR i: lanes0-15 M=i, lanes16-31 M=8+i; N = lane%16
#pragma unroll
  for (int mi = 0; mi < 2; ++mi) {
    const int mRow = mBase + waveM * 32 + mi * 16 + hi * 8;
#pragma unroll
    for (int ni = 0; ni < 4; ++ni) {
      const int nCol = nBase + waveN * 64 + ni * 16 + lm;
#pragma unroll
      for (int i = 0; i < 8; ++i)
        C[(size_t)(mRow + i) * D + nCol] = acc[mi][ni][i];
    }
  }
}

// ---------------------------------------------------------------------------
// launch
// ---------------------------------------------------------------------------
extern "C" void kernel_launch(void* const* d_in, const int* in_sizes, int n_in,
                              void* d_out, int out_size, void* d_ws, size_t ws_size,
                              hipStream_t stream) {
  const float* x = (const float*)d_in[0];
  const float* w[3] = {(const float*)d_in[1], (const float*)d_in[2], (const float*)d_in[3]};
  float* out = (float*)d_out;

  char* ws = (char*)d_ws;
  float* scales   = (float*)ws;                         // 3 floats
  float* partials = (float*)(ws + 256);                 // 3 * 512 floats
  _Float16* wq    = (_Float16*)(ws + 8192);             // 3 * 16M f16 (96 MB)
  _Float16* actH  = wq + 3 * WELEM;                     // 32M f16 (64 MB)

  // quantize the three weight matrices
  for (int i = 0; i < 3; ++i) {
    absmean_kernel<<<RED_BLOCKS, 256, 0, stream>>>(w[i], partials + i * RED_BLOCKS,
                                                   (int)WELEM);
    finalize_scale_kernel<<<1, 256, 0, stream>>>(partials + i * RED_BLOCKS, scales + i);
    ternarize_kernel<<<(int)(WELEM / 4 + 255) / 256, 256, 0, stream>>>(
        w[i], scales + i, wq + i * WELEM, (int)(WELEM / 4));
  }

  // three blocks: norm+silu -> f16, then WMMA GEMM -> fp32 (d_out as ping buffer)
  dim3 gemmGrid(D / TN, NTOK / TM);   // (32, 64)
  const float* cur = x;
  for (int i = 0; i < 3; ++i) {
    norm_silu_kernel<<<NTOK, 256, 0, stream>>>(cur, actH);
    gemm_tern_kernel<<<gemmGrid, 256, 0, stream>>>(actH, wq + i * WELEM, out);
    cur = out;
  }
}
